// Hybrid_44452911513784
// MI455X (gfx1250) — matrix operand order repaired and checked by reading, not compiled, not run on hardware
//
#include <hip/hip_runtime.h>

#define NEG_SLOPE 0.2f
#define BN_EPS 1e-5f

typedef __attribute__((ext_vector_type(16))) _Float16     v16h;
typedef __attribute__((ext_vector_type(8)))  _Float16     v8h;
typedef __attribute__((ext_vector_type(8)))  float        v8f;
typedef __attribute__((ext_vector_type(4)))  unsigned int v4u;
typedef __attribute__((ext_vector_type(8)))  int          v8i;
typedef __attribute__((ext_vector_type(4)))  int          v4i;

__device__ __forceinline__ float leakyf(float x) { return x > 0.f ? x : NEG_SLOPE * x; }
__device__ __forceinline__ float eluf(float x)   { return x > 0.f ? x : (__expf(x) - 1.f); }

__device__ __forceinline__ void atomicMaxF(float* addr, float val) {
  int* ai = reinterpret_cast<int*>(addr);
  int old = __float_as_int(*addr);
  while (__int_as_float(old) < val) {
    int assumed = old;
    old = atomicCAS(ai, assumed, __float_as_int(val));
    if (old == assumed) break;
  }
}

// ---------------------------------------------------------------------------
// WMMA GEMM on f16 activations/weights, f32 accumulate:
//   C[N x M] = act( (acc ? C : 0) + A16[N x K] @ Wt16[M x K]^T + bias )
// A16 is row-major f16. Wt16 is the WEIGHT TRANSPOSE (Wt[m][k] = W[k][m]), so
// one 16-column LDS tile == one contiguous 16*K-element span -> a single 1-D
// Tensor Data Mover descriptor moves it (wave 0 issues, TENSORcnt-waited).
// Block = 128 threads (4 waves); each wave owns one 16x16 C tile (64x16/block).
// K % 32 == 0, M % 16 == 0.
// ---------------------------------------------------------------------------
__global__ void gemm_wmma_kernel(const _Float16* __restrict__ A,
                                 const _Float16* __restrict__ Wt,
                                 const float* __restrict__ bias,
                                 float* __restrict__ C,
                                 int Nrows, int K, int M,
                                 int accFromC, int act)
{
  __shared__ _Float16 Wl[256 * 16];   // up to K=256
  const int tid  = threadIdx.x;
  const int lane = tid & 31;
  const int wave = tid >> 5;
  const int col0 = blockIdx.y * 16;
  const int row0 = blockIdx.x * 64 + wave * 16;

  // ---- Stage W tile (16 cols x K) into LDS: contiguous copy of 16*K halves.
#if __has_builtin(__builtin_amdgcn_tensor_load_to_lds)
  if (tid < 32) {
    const unsigned long long ga =
        (unsigned long long)(uintptr_t)(Wt + (size_t)col0 * K);
    const unsigned int lds  = (unsigned int)(uintptr_t)(&Wl[0]);  // low 32 = LDS addr
    const unsigned int td0  = (unsigned int)(16 * K);             // elements (f16)
    // D# group0: count=1 | lds_addr | global_addr[56:0] | type=2
    v4u g0 = { 1u, lds, (unsigned int)ga,
               (unsigned int)((ga >> 32) & 0x01FFFFFFull) | (2u << 30) };
    // D# group1: data_size=2B; tensor_dim0=td0; tensor_dim1=1; tile_dim0=td0;
    //            tile_dim1=1; tensor_dim0_stride=td0
    v8i g1 = { (int)(1u << 16),
               (int)((td0 & 0xFFFFu) << 16),
               (int)(1u << 16),
               (int)(((td0 & 0xFFFFu) << 16) | 1u),
               0,
               (int)td0,
               0, 0 };
    v4i g2 = { 0, 0, 0, 0 };
    v4i g3 = { 0, 0, 0, 0 };
#if __clang_major__ >= 23
    v8i gx = { 0, 0, 0, 0, 0, 0, 0, 0 };
    __builtin_amdgcn_tensor_load_to_lds(g0, g1, g2, g3, gx, 0);
#else
    __builtin_amdgcn_tensor_load_to_lds(g0, g1, g2, g3, 0);
#endif
    __builtin_amdgcn_s_wait_tensorcnt(0);
  }
  __syncthreads();
#else
  {
    const _Float16* wsrc = Wt + (size_t)col0 * K;
    for (int idx = tid; idx < 16 * K; idx += blockDim.x) Wl[idx] = wsrc[idx];
    __syncthreads();
  }
#endif

  const int m  = lane & 15;           // A-frag row / B,C column within tile
  const int hi = lane >> 4;           // lane half select
  const int row  = row0 + m;
  const int rowc = row < Nrows ? row : (Nrows - 1);   // clamp: EXEC stays all-1s
  const _Float16* arow = A + (size_t)rowc * K;
  const int col = col0 + m;

  v8f c;
  if (accFromC) {
#pragma unroll
    for (int v = 0; v < 8; ++v) {
      int rm = row0 + v + hi * 8;
      c[v] = (rm < Nrows) ? C[(size_t)rm * M + col] : 0.0f;
    }
  } else {
#pragma unroll
    for (int v = 0; v < 8; ++v) c[v] = 0.0f;
  }

  for (int k0 = 0; k0 < K; k0 += 32) {
    // A fragment (16-bit A 16x32 layout): halves 0..7 -> K = k0+hi*8 .. +7,
    // halves 8..15 -> K = k0+16+hi*8 .. +7. Two 16B loads, no conversion.
    const int koff = k0 + hi * 8;
    const v8h a0 = *reinterpret_cast<const v8h*>(arow + koff);
    const v8h a1 = *reinterpret_cast<const v8h*>(arow + koff + 16);
    v16h a;
#pragma unroll
    for (int h = 0; h < 8; ++h) { a[h] = a0[h]; a[h + 8] = a1[h]; }

    // B fragment (32x16): same K order as A: halves 0..7 -> K=k0+hi*8..+7, 8..15 -> K=k0+16+hi*8..+7.
    const int kb = k0 + hi * 8;
    const v8h b0 = *reinterpret_cast<const v8h*>(&Wl[(size_t)m * K + kb]);
    const v8h b1 = *reinterpret_cast<const v8h*>(&Wl[(size_t)m * K + kb + 16]);
    v16h b;
#pragma unroll
    for (int h = 0; h < 8; ++h) { b[h] = b0[h]; b[h + 8] = b1[h]; }

    c = __builtin_amdgcn_wmma_f32_16x16x32_f16(false, a, false, b, (short)0, c,
                                               false, false);
  }

  const float bv = bias ? bias[col] : 0.0f;
#pragma unroll
  for (int v = 0; v < 8; ++v) {
    int rm = row0 + v + hi * 8;
    if (rm < Nrows) {
      float val = c[v] + bv;
      if (act == 1) val = val > 0.f ? val : 0.f;
      C[(size_t)rm * M + col] = val;
    }
  }
}

// ---------------------------------------------------------------------------
// Precision / layout prep kernels
// ---------------------------------------------------------------------------
__global__ void f32_to_f16(const float* __restrict__ in, _Float16* __restrict__ out,
                           long long n) {
  long long i = (long long)blockIdx.x * blockDim.x + threadIdx.x;
  if (i < n) out[i] = (_Float16)in[i];
}

// Wt[m*K + k] = (f16) W[k*M + m]
__global__ void transpose_w(const float* __restrict__ W, _Float16* __restrict__ Wt,
                            int K, int M) {
  long long i = (long long)blockIdx.x * blockDim.x + threadIdx.x;
  if (i >= (long long)K * M) return;
  int k = (int)(i / M);
  int m = (int)(i - (long long)k * M);
  Wt[(size_t)m * K + k] = (_Float16)W[i];
}

// ---------------------------------------------------------------------------
// Graph / elementwise kernels
// ---------------------------------------------------------------------------
__global__ void zero_f32(float* p, long long n) {
  long long i = (long long)blockIdx.x * blockDim.x + threadIdx.x;
  if (i < n) p[i] = 0.0f;
}

__global__ void count_indeg(const int* __restrict__ dst, float* indeg, int E) {
  int i = blockIdx.x * blockDim.x + threadIdx.x;
  if (i < E) atomicAdd(&indeg[dst[i]], 1.0f);
}

__global__ void compute_dinv(const float* __restrict__ indeg, float* dinv, int Nn) {
  int i = blockIdx.x * blockDim.x + threadIdx.x;
  if (i < Nn) dinv[i] = rsqrtf(indeg[i] + 1.0f);   // self-loop included
}

__global__ void gcn_init(const float* __restrict__ h, const float* __restrict__ dinv,
                         float* __restrict__ out, int Nn, int F) {
  long long idx = (long long)blockIdx.x * blockDim.x + threadIdx.x;
  if (idx >= (long long)Nn * F) return;
  int n = (int)(idx / F);
  float di = dinv[n];
  out[idx] = h[idx] * di * di;
}

__global__ void gcn_edges(const float* __restrict__ h, const int* __restrict__ src,
                          const int* __restrict__ dst, const float* __restrict__ dinv,
                          float* __restrict__ out, int E, int F) {
  long long idx = (long long)blockIdx.x * blockDim.x + threadIdx.x;
  const int per = F >> 2;
  if (idx >= (long long)E * per) return;
  int e = (int)(idx / per);
  int f = (int)(idx - (long long)e * per) << 2;
  int s = src[e], d = dst[e];
  float norm = dinv[s] * dinv[d];
  const float4 hv = *reinterpret_cast<const float4*>(h + (size_t)s * F + f);
  float* o = out + (size_t)d * F + f;
  atomicAdd(o + 0, hv.x * norm);
  atomicAdd(o + 1, hv.y * norm);
  atomicAdd(o + 2, hv.z * norm);
  atomicAdd(o + 3, hv.w * norm);
}

// x = act(x + bias); optionally also emit f16 copy
__global__ void bias_act(float* x, const float* __restrict__ bias,
                         _Float16* __restrict__ xh, int Nn, int F, int act) {
  long long idx = (long long)blockIdx.x * blockDim.x + threadIdx.x;
  if (idx >= (long long)Nn * F) return;
  int c = (int)(idx % F);
  float v = x[idx] + (bias ? bias[c] : 0.0f);
  if (act == 1) v = v > 0.f ? v : 0.f;
  else if (act == 2) v = eluf(v);
  x[idx] = v;
  if (xh) xh[idx] = (_Float16)v;
}

__global__ void gat_attn(const float* __restrict__ h, const float* __restrict__ asrc,
                         const float* __restrict__ adst, float* a_s, float* a_d,
                         int Nn, int heads, int C) {
  long long idx = (long long)blockIdx.x * blockDim.x + threadIdx.x;
  if (idx >= (long long)Nn * heads) return;
  int n = (int)(idx / heads);
  int hd = (int)(idx - (long long)n * heads);
  const float* hv = h + (size_t)n * heads * C + (size_t)hd * C;
  const float* as = asrc + (size_t)hd * C;
  const float* ad = adst + (size_t)hd * C;
  float ss = 0.f, dd = 0.f;
  for (int c = 0; c < C; ++c) { float x = hv[c]; ss += x * as[c]; dd += x * ad[c]; }
  a_s[idx] = ss; a_d[idx] = dd;
}

__global__ void gat_emax_init(const float* __restrict__ a_s, const float* __restrict__ a_d,
                              float* emax, long long total) {
  long long i = (long long)blockIdx.x * blockDim.x + threadIdx.x;
  if (i < total) emax[i] = leakyf(a_s[i] + a_d[i]);   // self-loop edge
}

__global__ void gat_emax_edges(const float* __restrict__ a_s, const float* __restrict__ a_d,
                               const int* __restrict__ src, const int* __restrict__ dst,
                               float* emax, int E, int heads) {
  long long idx = (long long)blockIdx.x * blockDim.x + threadIdx.x;
  if (idx >= (long long)E * heads) return;
  int e = (int)(idx / heads);
  int hd = (int)(idx - (long long)e * heads);
  int s = src[e], d = dst[e];
  float v = leakyf(a_s[s * heads + hd] + a_d[d * heads + hd]);
  atomicMaxF(&emax[d * heads + hd], v);
}

__global__ void gat_init(const float* __restrict__ h, const float* __restrict__ a_s,
                         const float* __restrict__ a_d, const float* __restrict__ emax,
                         float* denom, float* out, int Nn, int heads, int C) {
  long long idx = (long long)blockIdx.x * blockDim.x + threadIdx.x;
  if (idx >= (long long)Nn * heads) return;
  float ex = __expf(leakyf(a_s[idx] + a_d[idx]) - emax[idx]);
  denom[idx] = ex;
  size_t base = (size_t)idx * C;
  const float* hv = h + base;
  float* ov = out + base;
  for (int c = 0; c < C; ++c) ov[c] = hv[c] * ex;
}

__global__ void gat_denom_edges(const float* __restrict__ a_s, const float* __restrict__ a_d,
                                const int* __restrict__ src, const int* __restrict__ dst,
                                const float* __restrict__ emax, float* denom, int E, int heads) {
  long long idx = (long long)blockIdx.x * blockDim.x + threadIdx.x;
  if (idx >= (long long)E * heads) return;
  int e = (int)(idx / heads);
  int hd = (int)(idx - (long long)e * heads);
  int s = src[e], d = dst[e];
  float ex = __expf(leakyf(a_s[s * heads + hd] + a_d[d * heads + hd]) - emax[d * heads + hd]);
  atomicAdd(&denom[d * heads + hd], ex);
}

__global__ void gat_out_edges(const float* __restrict__ h, const float* __restrict__ a_s,
                              const float* __restrict__ a_d, const int* __restrict__ src,
                              const int* __restrict__ dst, const float* __restrict__ emax,
                              float* out, int E, int heads, int C) {
  const int cq = C >> 2;
  const int per = heads * cq;
  long long idx = (long long)blockIdx.x * blockDim.x + threadIdx.x;
  if (idx >= (long long)E * per) return;
  int e = (int)(idx / per);
  int r = (int)(idx - (long long)e * per);
  int hd = r / cq;
  int f = (r - hd * cq) << 2;
  int s = src[e], d = dst[e];
  float ex = __expf(leakyf(a_s[s * heads + hd] + a_d[d * heads + hd]) - emax[d * heads + hd]);
  const float4 hv = *reinterpret_cast<const float4*>(h + (size_t)s * heads * C + (size_t)hd * C + f);
  float* o = out + (size_t)d * heads * C + (size_t)hd * C + f;
  atomicAdd(o + 0, hv.x * ex);
  atomicAdd(o + 1, hv.y * ex);
  atomicAdd(o + 2, hv.z * ex);
  atomicAdd(o + 3, hv.w * ex);
}

// out = elu(out/denom + bias); optional f16 copy
__global__ void gat_fin(float* out, const float* __restrict__ denom,
                        const float* __restrict__ bias, _Float16* __restrict__ outh,
                        int Nn, int heads, int C) {
  long long idx = (long long)blockIdx.x * blockDim.x + threadIdx.x;
  const int HC = heads * C;
  if (idx >= (long long)Nn * HC) return;
  int n = (int)(idx / HC);
  int hc = (int)(idx - (long long)n * HC);
  int hd = hc / C;
  float v = out[idx] / (denom[n * heads + hd] + 1e-16f) + bias[hc];
  v = eluf(v);
  out[idx] = v;
  if (outh) outh[idx] = (_Float16)v;
}

__global__ void sage_edges(const float* __restrict__ x, const int* __restrict__ src,
                           const int* __restrict__ dst, float* agg, int E, int F) {
  long long idx = (long long)blockIdx.x * blockDim.x + threadIdx.x;
  const int per = F >> 2;
  if (idx >= (long long)E * per) return;
  int e = (int)(idx / per);
  int f = (int)(idx - (long long)e * per) << 2;
  int s = src[e], d = dst[e];
  const float4 hv = *reinterpret_cast<const float4*>(x + (size_t)s * F + f);
  float* o = agg + (size_t)d * F + f;
  atomicAdd(o + 0, hv.x);
  atomicAdd(o + 1, hv.y);
  atomicAdd(o + 2, hv.z);
  atomicAdd(o + 3, hv.w);
}

// agg /= max(deg,1); emit f16 copy for the WMMA GEMM
__global__ void sage_div(float* agg, const float* __restrict__ indeg,
                         _Float16* __restrict__ aggh, int Nn, int F) {
  long long idx = (long long)blockIdx.x * blockDim.x + threadIdx.x;
  if (idx >= (long long)Nn * F) return;
  int n = (int)(idx / F);
  float v = agg[idx] / fmaxf(indeg[n], 1.0f);
  agg[idx] = v;
  aggh[idx] = (_Float16)v;
}

__global__ void bn_stats(const float* __restrict__ x, float* sum, float* sumsq, int Nn, int F) {
  long long idx = (long long)blockIdx.x * blockDim.x + threadIdx.x;
  if (idx >= (long long)Nn * F) return;
  int c = (int)(idx % F);
  float v = x[idx];
  atomicAdd(&sum[c], v);
  atomicAdd(&sumsq[c], v * v);
}

__global__ void bn_apply(float* x, const float* __restrict__ sum, const float* __restrict__ sumsq,
                         const float* __restrict__ g, const float* __restrict__ b, int Nn, int F) {
  long long idx = (long long)blockIdx.x * blockDim.x + threadIdx.x;
  if (idx >= (long long)Nn * F) return;
  int c = (int)(idx % F);
  float m = sum[c] / (float)Nn;
  float var = sumsq[c] / (float)Nn - m * m;
  x[idx] = (x[idx] - m) * rsqrtf(var + BN_EPS) * g[c] + b[c];
}

// concat -> f16 directly (feeds fc1 WMMA GEMM)
__global__ void concat3h(const float* __restrict__ g, const float* __restrict__ a,
                         const float* __restrict__ s, _Float16* cat, int Nn) {
  long long idx = (long long)blockIdx.x * blockDim.x + threadIdx.x;
  if (idx >= (long long)Nn * 192) return;
  int n = (int)(idx / 192);
  int c = (int)(idx - (long long)n * 192);
  float v;
  if (c < 64)       v = g[(size_t)n * 64 + c];
  else if (c < 128) v = a[(size_t)n * 64 + (c - 64)];
  else              v = s[(size_t)n * 64 + (c - 128)];
  cat[idx] = (_Float16)v;
}

__global__ void fc2_kernel(const float* __restrict__ f1, const float* __restrict__ w,
                           const float* __restrict__ b, float* out, int Nn) {
  int n = blockIdx.x * blockDim.x + threadIdx.x;
  if (n >= Nn) return;
  const float* v = f1 + (size_t)n * 64;
  float o0 = b[0], o1 = b[1];
  for (int k = 0; k < 64; ++k) {
    float x = v[k];
    o0 += x * w[k * 2 + 0];
    o1 += x * w[k * 2 + 1];
  }
  out[n * 2 + 0] = o0;
  out[n * 2 + 1] = o1;
}

// ---------------------------------------------------------------------------
extern "C" void kernel_launch(void* const* d_in, const int* in_sizes, int n_in,
                              void* d_out, int out_size, void* d_ws, size_t ws_size,
                              hipStream_t stream)
{
  (void)n_in; (void)out_size; (void)ws_size;
  const float* x        = (const float*)d_in[0];
  const int*   ei       = (const int*)  d_in[1];
  const float* gcn1_w   = (const float*)d_in[2];
  const float* gcn1_b   = (const float*)d_in[3];
  const float* gcn2_w   = (const float*)d_in[4];
  const float* gcn2_b   = (const float*)d_in[5];
  const float* gat1_w   = (const float*)d_in[6];
  const float* gat1_as  = (const float*)d_in[7];
  const float* gat1_ad  = (const float*)d_in[8];
  const float* gat1_b   = (const float*)d_in[9];
  const float* gat2_w   = (const float*)d_in[10];
  const float* gat2_as  = (const float*)d_in[11];
  const float* gat2_ad  = (const float*)d_in[12];
  const float* gat2_b   = (const float*)d_in[13];
  const float* sage1_wl = (const float*)d_in[14];
  const float* sage1_bl = (const float*)d_in[15];
  const float* sage1_wr = (const float*)d_in[16];
  const float* sage2_wl = (const float*)d_in[17];
  const float* sage2_bl = (const float*)d_in[18];
  const float* sage2_wr = (const float*)d_in[19];
  const float* bn_gcn_g = (const float*)d_in[20];
  const float* bn_gcn_b = (const float*)d_in[21];
  const float* bn_gat_g = (const float*)d_in[22];
  const float* bn_gat_b = (const float*)d_in[23];
  const float* bn_sg_g  = (const float*)d_in[24];
  const float* bn_sg_b  = (const float*)d_in[25];
  const float* fc1_w    = (const float*)d_in[26];
  const float* fc1_b    = (const float*)d_in[27];
  const float* fc2_w    = (const float*)d_in[28];
  const float* fc2_b    = (const float*)d_in[29];

  const int N = in_sizes[0] / 128;
  const int E = in_sizes[1] / 2;
  const int* src = ei;
  const int* dst = ei + E;

  char* base = (char*)d_ws;
  size_t off = 0;
  auto allocf = [&](size_t n) { float* p = (float*)(base + off);
                                off += ((n * 4 + 15) & ~(size_t)15); return p; };
  auto alloch = [&](size_t n) { _Float16* p = (_Float16*)(base + off);
                                off += ((n * 2 + 15) & ~(size_t)15); return p; };
  // f32 buffers
  float* indeg = allocf(N);
  float* dinv  = allocf(N);
  float* t0    = allocf((size_t)N * 256);
  float* g1    = allocf((size_t)N * 64);
  float* g2    = allocf((size_t)N * 64);
  float* a1    = allocf((size_t)N * 256);
  float* a2    = allocf((size_t)N * 64);
  float* s1    = allocf((size_t)N * 64);
  float* s2    = allocf((size_t)N * 64);
  float* aggx  = allocf((size_t)N * 128);
  float* aggs  = allocf((size_t)N * 64);
  float* asb   = allocf((size_t)N * 4);
  float* adb   = allocf((size_t)N * 4);
  float* emax  = allocf((size_t)N * 4);
  float* denom = allocf((size_t)N * 4);
  float* f1    = allocf((size_t)N * 64);
  float* bnsum = allocf(64);
  float* bnsq  = allocf(64);
  // f16 activation buffers
  _Float16* xh    = alloch((size_t)N * 128);
  _Float16* g1h   = alloch((size_t)N * 64);
  _Float16* a1h   = alloch((size_t)N * 256);
  _Float16* aggxh = alloch((size_t)N * 128);
  _Float16* aggsh = alloch((size_t)N * 64);
  _Float16* s1h   = alloch((size_t)N * 64);
  _Float16* cath  = alloch((size_t)N * 192);
  // f16 transposed weights Wt[M][K]
  _Float16* gcn1_wt  = alloch(128 * 64);
  _Float16* gcn2_wt  = alloch(64 * 64);
  _Float16* gat1_wt  = alloch(128 * 256);
  _Float16* gat2_wt  = alloch(256 * 64);
  _Float16* sage1_wlt = alloch(128 * 64);
  _Float16* sage1_wrt = alloch(128 * 64);
  _Float16* sage2_wlt = alloch(64 * 64);
  _Float16* sage2_wrt = alloch(64 * 64);
  _Float16* fc1_wt   = alloch(192 * 64);

  auto nb = [](long long total) { return (int)((total + 255) / 256); };
  auto gemm = [&](const _Float16* A, const _Float16* Wt, const float* bias, float* C,
                  int K, int M, int acc, int act) {
    dim3 grid((N + 63) / 64, M / 16);
    gemm_wmma_kernel<<<grid, 128, 0, stream>>>(A, Wt, bias, C, N, K, M, acc, act);
  };
  auto bn = [&](float* buf, const float* g, const float* b) {
    zero_f32<<<1, 256, 0, stream>>>(bnsum, 128);   // bnsum+bnsq contiguous
    bn_stats<<<nb((long long)N * 64), 256, 0, stream>>>(buf, bnsum, bnsq, N, 64);
    bn_apply<<<nb((long long)N * 64), 256, 0, stream>>>(buf, bnsum, bnsq, g, b, N, 64);
  };
  auto prep_w = [&](const float* W, _Float16* Wt, int K, int M) {
    transpose_w<<<nb((long long)K * M), 256, 0, stream>>>(W, Wt, K, M);
  };

  // ---- weight layout/precision prep (tiny) ----
  prep_w(gcn1_w, gcn1_wt, 128, 64);
  prep_w(gcn2_w, gcn2_wt, 64, 64);
  prep_w(gat1_w, gat1_wt, 128, 256);
  prep_w(gat2_w, gat2_wt, 256, 64);
  prep_w(sage1_wl, sage1_wlt, 128, 64);
  prep_w(sage1_wr, sage1_wrt, 128, 64);
  prep_w(sage2_wl, sage2_wlt, 64, 64);
  prep_w(sage2_wr, sage2_wrt, 64, 64);
  prep_w(fc1_w, fc1_wt, 192, 64);
  f32_to_f16<<<nb((long long)N * 128), 256, 0, stream>>>(x, xh, (long long)N * 128);

  // ---- degrees ----
  zero_f32<<<nb(N), 256, 0, stream>>>(indeg, N);
  count_indeg<<<nb(E), 256, 0, stream>>>(dst, indeg, E);
  compute_dinv<<<nb(N), 256, 0, stream>>>(indeg, dinv, N);

  // ---- GCN branch ----
  gemm(xh, gcn1_wt, nullptr, t0, 128, 64, 0, 0);
  gcn_init<<<nb((long long)N * 64), 256, 0, stream>>>(t0, dinv, g1, N, 64);
  gcn_edges<<<nb((long long)E * 16), 256, 0, stream>>>(t0, src, dst, dinv, g1, E, 64);
  bias_act<<<nb((long long)N * 64), 256, 0, stream>>>(g1, gcn1_b, g1h, N, 64, 1);

  gemm(g1h, gcn2_wt, nullptr, t0, 64, 64, 0, 0);
  gcn_init<<<nb((long long)N * 64), 256, 0, stream>>>(t0, dinv, g2, N, 64);
  gcn_edges<<<nb((long long)E * 16), 256, 0, stream>>>(t0, src, dst, dinv, g2, E, 64);
  bias_act<<<nb((long long)N * 64), 256, 0, stream>>>(g2, gcn2_b, (_Float16*)nullptr, N, 64, 1);
  bn(g2, bn_gcn_g, bn_gcn_b);

  // ---- GAT branch, layer 1 (heads=4, C=64, concat) ----
  gemm(xh, gat1_wt, nullptr, t0, 128, 256, 0, 0);
  gat_attn<<<nb((long long)N * 4), 256, 0, stream>>>(t0, gat1_as, gat1_ad, asb, adb, N, 4, 64);
  gat_emax_init<<<nb((long long)N * 4), 256, 0, stream>>>(asb, adb, emax, (long long)N * 4);
  gat_emax_edges<<<nb((long long)E * 4), 256, 0, stream>>>(asb, adb, src, dst, emax, E, 4);
  gat_init<<<nb((long long)N * 4), 256, 0, stream>>>(t0, asb, adb, emax, denom, a1, N, 4, 64);
  gat_denom_edges<<<nb((long long)E * 4), 256, 0, stream>>>(asb, adb, src, dst, emax, denom, E, 4);
  gat_out_edges<<<nb((long long)E * 64), 256, 0, stream>>>(t0, asb, adb, src, dst, emax, a1, E, 4, 64);
  gat_fin<<<nb((long long)N * 256), 256, 0, stream>>>(a1, denom, gat1_b, a1h, N, 4, 64);

  // ---- GAT branch, layer 2 (heads=1, C=64, mean==identity) ----
  gemm(a1h, gat2_wt, nullptr, t0, 256, 64, 0, 0);
  gat_attn<<<nb(N), 256, 0, stream>>>(t0, gat2_as, gat2_ad, asb, adb, N, 1, 64);
  gat_emax_init<<<nb(N), 256, 0, stream>>>(asb, adb, emax, (long long)N);
  gat_emax_edges<<<nb(E), 256, 0, stream>>>(asb, adb, src, dst, emax, E, 1);
  gat_init<<<nb(N), 256, 0, stream>>>(t0, asb, adb, emax, denom, a2, N, 1, 64);
  gat_denom_edges<<<nb(E), 256, 0, stream>>>(asb, adb, src, dst, emax, denom, E, 1);
  gat_out_edges<<<nb((long long)E * 16), 256, 0, stream>>>(t0, asb, adb, src, dst, emax, a2, E, 1, 64);
  gat_fin<<<nb((long long)N * 64), 256, 0, stream>>>(a2, denom, gat2_b, (_Float16*)nullptr, N, 1, 64);
  bn(a2, bn_gat_g, bn_gat_b);

  // ---- SAGE branch ----
  zero_f32<<<nb((long long)N * 128), 256, 0, stream>>>(aggx, (long long)N * 128);
  sage_edges<<<nb((long long)E * 32), 256, 0, stream>>>(x, src, dst, aggx, E, 128);
  sage_div<<<nb((long long)N * 128), 256, 0, stream>>>(aggx, indeg, aggxh, N, 128);
  gemm(aggxh, sage1_wlt, sage1_bl, s1, 128, 64, 0, 0);
  gemm(xh, sage1_wrt, nullptr, s1, 128, 64, 1, 1);        // s1 = relu(s1 + x@wr)
  f32_to_f16<<<nb((long long)N * 64), 256, 0, stream>>>(s1, s1h, (long long)N * 64);

  zero_f32<<<nb((long long)N * 64), 256, 0, stream>>>(aggs, (long long)N * 64);
  sage_edges<<<nb((long long)E * 16), 256, 0, stream>>>(s1, src, dst, aggs, E, 64);
  sage_div<<<nb((long long)N * 64), 256, 0, stream>>>(aggs, indeg, aggsh, N, 64);
  gemm(aggsh, sage2_wlt, sage2_bl, s2, 64, 64, 0, 0);
  gemm(s1h, sage2_wrt, nullptr, s2, 64, 64, 1, 1);
  bn(s2, bn_sg_g, bn_sg_b);

  // ---- fusion head ----
  concat3h<<<nb((long long)N * 192), 256, 0, stream>>>(g2, a2, s2, cath, N);
  gemm(cath, fc1_wt, fc1_b, f1, 192, 64, 0, 1);
  fc2_kernel<<<nb(N), 256, 0, stream>>>(f1, fc2_w, fc2_b, (float*)d_out, N);
}
